// TBSA_32014686224566
// MI455X (gfx1250) — matrix operand order repaired and checked by reading, not compiled, not run on hardware
//
#include <hip/hip_runtime.h>
#include <hip/hip_bf16.h>
#include <cstdint>
#include <cstddef>

// ---------------------------------------------------------------------------
// Types for CDNA5 WMMA / TDM (gfx1250, wave32)
// ---------------------------------------------------------------------------
typedef __attribute__((ext_vector_type(16))) __bf16        v16bf;
typedef __attribute__((ext_vector_type(8)))  float         v8f;
typedef __attribute__((ext_vector_type(8)))  unsigned int  v8u;
typedef __attribute__((ext_vector_type(4)))  unsigned int  v4u;
typedef __attribute__((ext_vector_type(4)))  unsigned int  u32x4_t;
typedef __attribute__((ext_vector_type(8)))  int           i32x8_t;
typedef __attribute__((ext_vector_type(4)))  int           i32x4_t;

#define B_   128
#define T_   128
#define H_   256
#define G_   1024   // 4H
#define E_   300
#define EP_  320    // E padded to multiple of 32 (WMMA K)
#define D2_  512    // 2H

// TDM builtin arity differs across toolchains: therock-10.0 headers ship the
// gfx1250 TDM header and use the 6-arg form; ROCm 7.2 clang-22 is 5-arg.
#if defined(__has_include)
#  if __has_include(<hip/amd_detail/amd_gfx1250_TDM.h>)
#    define TBSA_TDM_6ARG 1
#  else
#    define TBSA_TDM_6ARG 0
#  endif
#else
#  define TBSA_TDM_6ARG 0
#endif

__device__ __forceinline__ unsigned short f2bf(float f) {
  unsigned int u = __builtin_bit_cast(unsigned int, f);
  u += 0x7FFFu + ((u >> 16) & 1u);     // round-to-nearest-even
  return (unsigned short)(u >> 16);
}
__device__ __forceinline__ float sigm(float x) { return 1.0f / (1.0f + __expf(-x)); }

// D = A(16x32 bf16) * B(32x16 bf16) + C(16x16 f32)
__device__ __forceinline__ v8f wmma_bf16(v8u a8, v8u b8, v8f c) {
  v16bf a = __builtin_bit_cast(v16bf, a8);
  v16bf b = __builtin_bit_cast(v16bf, b8);
  return __builtin_amdgcn_wmma_f32_16x16x32_bf16(
      /*neg_a=*/false, a, /*neg_b=*/false, b,
      /*c_mod=*/(short)0, c, /*reuse_a=*/false, /*reuse_b=*/false);
}

// ---------------------------------------------------------------------------
// TDM: async-load a 2D f32 tile (16 rows x 1024 cols, row stride 1024 elems)
// from global memory into LDS at byte offset lds_off. D# per ISA 8.3/8.4:
//   group0: count=1 | lds_addr | global_addr[56:0] | type=2
//   group1: data_size=4B | tensor 1024x16 | tile 1024x16 | dim0_stride=1024
// Completion tracked with TENSORcnt.
// ---------------------------------------------------------------------------
__device__ __forceinline__ void tdm_load_xg(unsigned lds_off, const float* gptr) {
  const unsigned long long ga = (unsigned long long)(uintptr_t)gptr;
  u32x4_t g0;
  g0[0] = 1u;                                                    // count=1
  g0[1] = lds_off;                                               // lds_addr
  g0[2] = (unsigned)ga;                                          // global_addr lo
  g0[3] = (unsigned)((ga >> 32) & 0x01FFFFFFull) | 0x80000000u;  // addr hi | type=2
  i32x8_t g1;
  g1[0] = (int)(2u << 16);        // workgroup_mask=0, data_size=2 (4 bytes)
  g1[1] = (int)(1024u << 16);     // tensor_dim0[15:0]=1024 (bits 63:48)
  g1[2] = (int)(16u << 16);       // tensor_dim0 hi=0 | tensor_dim1=16
  g1[3] = (int)(1024u << 16);     // tensor_dim1 hi=0 | tile_dim0=1024
  g1[4] = 16;                     // tile_dim1=16, tile_dim2=0
  g1[5] = 1024;                   // tensor_dim0_stride lo32 = 1024 elems
  g1[6] = 0;                      // stride hi | tensor_dim1_stride lo
  g1[7] = 0;
  i32x4_t gz = {0, 0, 0, 0};
#if TBSA_TDM_6ARG
  i32x8_t gz8 = {0, 0, 0, 0, 0, 0, 0, 0};
  __builtin_amdgcn_tensor_load_to_lds(g0, g1, gz, gz, gz8, 0);
#else
  __builtin_amdgcn_tensor_load_to_lds(g0, g1, gz, gz, 0);
#endif
}

// ---------------------------------------------------------------------------
// f32 -> bf16 convert with optional K padding (rows x ksrc -> rows x kdst)
// ---------------------------------------------------------------------------
__global__ void k_cvt_pad(const float* __restrict__ src, unsigned short* __restrict__ dst,
                          int rows, int ks, int kd) {
  const int i = blockIdx.x * blockDim.x + threadIdx.x;
  if (i >= rows * kd) return;
  const int r = i / kd, k = i - r * kd;
  dst[i] = (k < ks) ? f2bf(src[(size_t)r * ks + k]) : (unsigned short)0;
}

// ---------------------------------------------------------------------------
// Embedding gather: ids[B*T] -> ebuf[B*T][EP_] bf16 (zero-padded K)
// ---------------------------------------------------------------------------
__global__ void k_embed(const int* __restrict__ ids, const float* __restrict__ emb,
                        unsigned short* __restrict__ out) {
  const int row = blockIdx.x;
  const int id  = ids[row];
  for (int k = threadIdx.x; k < EP_; k += blockDim.x)
    out[(size_t)row * EP_ + k] = (k < E_) ? f2bf(emb[(size_t)id * E_ + k]) : (unsigned short)0;
}

// ---------------------------------------------------------------------------
// out f32 = A[M][K]bf16 @ W[N][K]bf16^T + bias[N]; one wave per 16x16 tile.
// Two-stage software pipeline: distinct fragment register sets so the wait
// before each WMMA covers only its own loads (partial s_wait_loadcnt) while
// the other stage's loads stay in flight. K/32 is even for all call sites.
// swap_bt=1 writes rows transposed from [b*T+t] to [t*B+b] (for xg layout).
// ---------------------------------------------------------------------------
__global__ __launch_bounds__(256) void k_gemm_bf16(
    const unsigned short* __restrict__ A, const unsigned short* __restrict__ W,
    const float* __restrict__ bias, float* __restrict__ out,
    int M, int N, int K, int swap_bt) {
  const int wid = blockIdx.x * 8 + (threadIdx.x >> 5);
  const int tn  = N >> 4;
  const int mt  = wid / tn;
  const int nt  = wid - mt * tn;
  if (mt * 16 >= M) return;
  const int lane  = threadIdx.x & 31;
  const int col16 = lane & 15;
  const int khalf = lane >> 4;
  const int m0 = mt * 16, n0 = nt * 16;
  const unsigned short* pa0 = A + (size_t)(m0 + col16) * K + khalf * 16;
  const unsigned short* pb0 = W + (size_t)(n0 + col16) * K + khalf * 16;

  auto ldfrag = [&](int k0, v8u& a, v8u& b) {
    const v4u alo = *(const v4u*)(pa0 + k0);
    const v4u ahi = *(const v4u*)(pa0 + k0 + 8);
#pragma unroll
    for (int q = 0; q < 4; ++q) { a[q] = alo[q]; a[q + 4] = ahi[q]; }
    b = *(const v8u*)(pb0 + k0);
  };

  v8f acc = {};
  v8u aA, bA, aB, bB;
  ldfrag(0, aA, bA);
  for (int k0 = 0; k0 < K; k0 += 64) {
    ldfrag(k0 + 32, aB, bB);                       // stage B loads in flight
    if (k0 + 64 < K) __builtin_prefetch(pb0 + k0 + 64, 0, 1);   // global_prefetch_b8
    acc = wmma_bf16(aA, bA, acc);                  // waits only on stage A
    if (k0 + 64 < K) ldfrag(k0 + 64, aA, bA);      // stage A loads in flight
    acc = wmma_bf16(aB, bB, acc);                  // waits only on stage B
  }

  const float bv = bias ? bias[n0 + col16] : 0.0f;
#pragma unroll
  for (int r = 0; r < 8; ++r) {
    const int m = m0 + khalf * 8 + r;
    const int orow = swap_bt ? (((m & (T_ - 1)) << 7) | (m >> 7)) : m;  // [t][b]
    out[(size_t)orow * N + n0 + col16] = acc[r] + bv;
  }
}

// ---------------------------------------------------------------------------
// LSTM recurrence, batch-sliced: grid = 16 blocks (dir x 8 slices of 16 rows).
// Per step: g = xg[t] + h @ Whh^T (WMMA; h in LDS bf16, A-frag shared across
// 8 WMMAs per K-step). xg[t] slices are TDM-prefetched one step ahead into a
// double-buffered LDS stage (tensor_load_to_lds + s_wait_tensorcnt).
// Dynamic LDS: 8KB h + 2x64KB xg = 136KB.
// ---------------------------------------------------------------------------
#define SCAN_LDS_BYTES (16 * H_ * 2 + 2 * 16 * G_ * 4)

__global__ __launch_bounds__(256) void k_lstm_scan(
    const float* __restrict__ xg_base,            // [2][T][B][4H] f32 (bias folded)
    const unsigned short* __restrict__ whh_base,  // [2][4H][H] bf16
    const float* __restrict__ h0,                 // [2][B][H]
    const float* __restrict__ c0,                 // [2][B][H]
    float* __restrict__ cst,                      // [2][B][H] scratch cell state
    float* __restrict__ obase)                    // [B][T][2H], dir selects half
{
  extern __shared__ char smem[];
  unsigned short* hls = (unsigned short*)smem;              // 16*256 bf16 = 8KB
  float*          xls = (float*)(smem + 16 * H_ * 2);       // 2 x 16*1024 f32
  const unsigned  lds_xg0 = (unsigned)(uintptr_t)(smem + 16 * H_ * 2);

  const int dir = blockIdx.x >> 3;
  const int b0  = (blockIdx.x & 7) * 16;
  const float* xg = xg_base + (size_t)dir * T_ * B_ * G_;
  const unsigned short* whh = whh_base + (size_t)dir * G_ * H_;
  float* cs = cst + (size_t)dir * B_ * H_;
  const int tid = threadIdx.x;

  for (int i = tid; i < 16 * H_; i += 256) {
    const int gi = (b0 + (i >> 8)) * H_ + (i & (H_ - 1));
    hls[i] = f2bf(h0[(size_t)dir * B_ * H_ + gi]);
    cs[gi] = c0[(size_t)dir * B_ * H_ + gi];
  }
  const int wave  = tid >> 5;
  const int lane  = tid & 31;
  const int col16 = lane & 15;
  const int khalf = lane >> 4;
  const unsigned short* pa0 = hls + col16 * H_ + khalf * 16;   // A frag base (LDS)

  // Preload xg slice for step 0 into buffer 0 (wave 0 issues the DMA).
  if (wave == 0) {
    const int t0 = dir ? (T_ - 1) : 0;
    tdm_load_xg(lds_xg0, xg + (size_t)t0 * B_ * G_ + (size_t)b0 * G_);
  }
  __syncthreads();

  for (int s = 0; s < T_; ++s) {
    const int t = dir ? (T_ - 1 - s) : s;
    // Prefetch next step's xg slice into the other buffer.
    if (wave == 0 && s + 1 < T_) {
      const int tn = dir ? (T_ - 2 - s) : (s + 1);
      tdm_load_xg(lds_xg0 + ((s + 1) & 1) * (16 * G_ * 4),
                  xg + (size_t)tn * B_ * G_ + (size_t)b0 * G_);
    }
    // Accumulate h @ Whh^T: A fragment shared by 8 WMMAs per K-step.
    v8f acc[2][4] = {};
#pragma unroll
    for (int k0 = 0; k0 < H_; k0 += 32) {
      const v4u alo = *(const v4u*)(pa0 + k0);        // ds_load_b128
      const v4u ahi = *(const v4u*)(pa0 + k0 + 8);
      v8u a;
#pragma unroll
      for (int q = 0; q < 4; ++q) { a[q] = alo[q]; a[q + 4] = ahi[q]; }
#pragma unroll
      for (int p = 0; p < 2; ++p) {
        const int j0 = (wave * 2 + p) * 16;
#pragma unroll
        for (int g = 0; g < 4; ++g) {
          const v8u b = *(const v8u*)(whh + (size_t)(g * H_ + j0 + col16) * H_ + k0 + khalf * 16);
          acc[p][g] = wmma_bf16(a, b, acc[p][g]);
        }
      }
    }
    // Current step's xg tile must be resident (allow the newest DMA in flight).
    if (s + 1 < T_) __builtin_amdgcn_s_wait_tensorcnt(1);
    else            __builtin_amdgcn_s_wait_tensorcnt(0);
    __syncthreads();   // all reads of old h done + xg[s] visible to all waves

    const float* xt = xls + (size_t)(s & 1) * (16 * G_);
#pragma unroll
    for (int p = 0; p < 2; ++p) {
      const int col = (wave * 2 + p) * 16 + col16;
#pragma unroll
      for (int r = 0; r < 8; ++r) {
        const int lrow = khalf * 8 + r;
        const int grow = b0 + lrow;
        const float gi = acc[p][0][r] + xt[lrow * G_ + 0 * H_ + col];
        const float gf = acc[p][1][r] + xt[lrow * G_ + 1 * H_ + col];
        const float gg = acc[p][2][r] + xt[lrow * G_ + 2 * H_ + col];
        const float go = acc[p][3][r] + xt[lrow * G_ + 3 * H_ + col];
        const size_t cidx = (size_t)grow * H_ + col;
        const float cv = sigm(gf) * cs[cidx] + sigm(gi) * tanhf(gg);
        cs[cidx] = cv;
        const float hv = sigm(go) * tanhf(cv);
        obase[((size_t)grow * T_ + t) * D2_ + dir * H_ + col] = hv;
        hls[lrow * H_ + col] = f2bf(hv);     // safe: all reads completed
      }
    }
    __syncthreads();   // new h visible
  }
}

// ---------------------------------------------------------------------------
// gt[b,t] = sigmoid(o1[b,t,:] . gt_W + gt_b); one wave per token
// ---------------------------------------------------------------------------
__global__ __launch_bounds__(256) void k_gt(const float* __restrict__ o1,
                                            const float* __restrict__ gtW,
                                            const float* __restrict__ gtb,
                                            float* __restrict__ gt) {
  const int tok  = blockIdx.x * 8 + (threadIdx.x >> 5);
  const int lane = threadIdx.x & 31;
  const float* row = o1 + (size_t)tok * D2_;
  float s = 0.f;
  for (int k = lane; k < D2_; k += 32) s += row[k] * gtW[k];
#pragma unroll
  for (int m = 16; m; m >>= 1) s += __shfl_xor(s, m, 32);
  if (lane == 0) gt[tok] = sigm(s + gtb[0]);
}

// ---------------------------------------------------------------------------
// hs2 gated scan over T: thread per (b, channel)
// ---------------------------------------------------------------------------
__global__ void k_hs2(const float* __restrict__ o2, const float* __restrict__ gt,
                      float* __restrict__ hs2) {
  const int i = blockIdx.x * blockDim.x + threadIdx.x;
  if (i >= B_ * D2_) return;
  const int b = i >> 9, ch = i & (D2_ - 1);
  const float* ob = o2  + (size_t)b * T_ * D2_ + ch;
  float*       hb = hs2 + (size_t)b * T_ * D2_ + ch;
  float prev = ob[0];
  hb[0] = prev;
  for (int t = 1; t < T_; ++t) {
    const float g = gt[b * T_ + t];
    prev = g * ob[(size_t)t * D2_] + (1.f - g) * prev;
    hb[(size_t)t * D2_] = prev;
  }
}

// ---------------------------------------------------------------------------
// Per-token heads: softmaxes, transition mix, NLL partials, argmax.
// ---------------------------------------------------------------------------
__global__ __launch_bounds__(256) void k_head(
    const float* __restrict__ o1,  const float* __restrict__ hs2,
    const float* __restrict__ c1W, const float* __restrict__ c1b,
    const float* __restrict__ c2W, const float* __restrict__ c2b,
    const float* __restrict__ c3W, const float* __restrict__ c3b,
    const float* __restrict__ trans,
    const float* __restrict__ yop, const float* __restrict__ yb,
    const float* __restrict__ yoe,
    float* __restrict__ predict, float* __restrict__ partial) {
  const int tok  = blockIdx.x * 8 + (threadIdx.x >> 5);
  const int lane = threadIdx.x & 31;
  const float* r1 = o1  + (size_t)tok * D2_;
  const float* r2 = hs2 + (size_t)tok * D2_;
  float dots[20];
#pragma unroll 1
  for (int c = 0; c < 20; ++c) {
    const float* w;
    const float* x;
    if (c < 5)       { w = c1W + c * D2_;        x = r1; }
    else if (c < 18) { w = c2W + (c - 5) * D2_;  x = r2; }
    else             { w = c3W + (c - 18) * D2_; x = r1; }
    float s = 0.f;
    for (int k = lane; k < D2_; k += 32) s += x[k] * w[k];
#pragma unroll
    for (int m = 16; m; m >>= 1) s += __shfl_xor(s, m, 32);
    dots[c] = s;
  }
  if (lane == 0) {
    float szt[5], szs[13], soe[2], szs2[13];
    float mx, sum;
    mx = -3.0e30f; sum = 0.f;
    for (int c = 0; c < 5;  ++c) { szt[c] = dots[c] + c1b[c];      mx = fmaxf(mx, szt[c]); }
    for (int c = 0; c < 5;  ++c) { szt[c] = __expf(szt[c] - mx);   sum += szt[c]; }
    for (int c = 0; c < 5;  ++c) szt[c] /= sum;
    mx = -3.0e30f; sum = 0.f;
    for (int c = 0; c < 13; ++c) { szs[c] = dots[5 + c] + c2b[c];  mx = fmaxf(mx, szs[c]); }
    for (int c = 0; c < 13; ++c) { szs[c] = __expf(szs[c] - mx);   sum += szs[c]; }
    for (int c = 0; c < 13; ++c) szs[c] /= sum;
    mx = -3.0e30f; sum = 0.f;
    for (int c = 0; c < 2;  ++c) { soe[c] = dots[18 + c] + c3b[c]; mx = fmaxf(mx, soe[c]); }
    for (int c = 0; c < 2;  ++c) { soe[c] = __expf(soe[c] - mx);   sum += soe[c]; }
    for (int c = 0; c < 2;  ++c) soe[c] /= sum;
    float at = 0.f;
    for (int c = 0; c < 5; ++c) at += szt[c] * szt[c];
    at *= 0.5f;
    for (int k = 0; k < 13; ++k) {
      float s = 0.f;
      for (int j = 0; j < 5; ++j) s += szt[j] * trans[j * 13 + k];
      szs2[k] = s;
    }
    float part = 0.f, best = -3.0e30f;
    int bi = 0;
    for (int k = 0; k < 13; ++k) {
      const float v = at * szs2[k] + (1.f - at) * szs[k];
      part += yop[(size_t)tok * 13 + k] * __logf(v);
      if (v > best) { best = v; bi = k; }
    }
    for (int c = 0; c < 5; ++c) part += yb[(size_t)tok * 5 + c]  * __logf(szt[c]);
    for (int c = 0; c < 2; ++c) part += yoe[(size_t)tok * 2 + c] * __logf(soe[c]);
    predict[tok] = (float)bi;
    partial[tok] = part;
  }
}

// Deterministic loss reduction: loss = -(1/T) * sum(partial)
__global__ __launch_bounds__(256) void k_loss(const float* __restrict__ partial,
                                              float* __restrict__ out_loss) {
  __shared__ float sm[256];
  float s = 0.f;
  for (int i = threadIdx.x; i < B_ * T_; i += 256) s += partial[i];
  sm[threadIdx.x] = s;
  __syncthreads();
  for (int m = 128; m; m >>= 1) {
    if (threadIdx.x < m) sm[threadIdx.x] += sm[threadIdx.x + m];
    __syncthreads();
  }
  if (threadIdx.x == 0) *out_loss = -sm[0] / (float)T_;
}

// ---------------------------------------------------------------------------
extern "C" void kernel_launch(void* const* d_in, const int* in_sizes, int n_in,
                              void* d_out, int out_size, void* d_ws, size_t ws_size,
                              hipStream_t stream) {
  (void)in_sizes; (void)n_in; (void)out_size; (void)ws_size;
  const int*   ids   = (const int*)d_in[0];
  const float* yop   = (const float*)d_in[1];
  const float* yb    = (const float*)d_in[2];
  const float* yoe   = (const float*)d_in[3];
  const float* wemb  = (const float*)d_in[4];
  const float* trans = (const float*)d_in[5];
  const float* l1Wih[2] = {(const float*)d_in[6],  (const float*)d_in[9]};
  const float* l1Whh[2] = {(const float*)d_in[7],  (const float*)d_in[10]};
  const float* l1b[2]   = {(const float*)d_in[8],  (const float*)d_in[11]};
  const float* l2Wih[2] = {(const float*)d_in[12], (const float*)d_in[15]};
  const float* l2Whh[2] = {(const float*)d_in[13], (const float*)d_in[16]};
  const float* l2b[2]   = {(const float*)d_in[14], (const float*)d_in[17]};
  const float* gtW = (const float*)d_in[18];
  const float* gtb = (const float*)d_in[19];
  const float* c1W = (const float*)d_in[20];
  const float* c1b = (const float*)d_in[21];
  const float* c2W = (const float*)d_in[22];
  const float* c2b = (const float*)d_in[23];
  const float* c3W = (const float*)d_in[24];
  const float* c3b = (const float*)d_in[25];
  const float* h01 = (const float*)d_in[26];
  const float* c01 = (const float*)d_in[27];
  const float* h02 = (const float*)d_in[28];
  const float* c02 = (const float*)d_in[29];

  char* ws = (char*)d_ws;
  size_t off = 0;
  auto alloc = [&](size_t bytes) -> char* {
    char* p = ws + off;
    off += (bytes + 255) & ~(size_t)255;
    return p;
  };
  unsigned short* ebuf = (unsigned short*)alloc((size_t)B_ * T_ * EP_ * 2);
  unsigned short* w1ih = (unsigned short*)alloc((size_t)2 * G_ * EP_ * 2);
  unsigned short* w1hh = (unsigned short*)alloc((size_t)2 * G_ * H_  * 2);
  unsigned short* w2ih = (unsigned short*)alloc((size_t)2 * G_ * D2_ * 2);
  unsigned short* w2hh = (unsigned short*)alloc((size_t)2 * G_ * H_  * 2);
  float*          xg1  = (float*)alloc((size_t)2 * B_ * T_ * G_ * 4);  // [2][T][B][4H]
  float*          xg2  = (float*)alloc((size_t)2 * B_ * T_ * G_ * 4);  // [2][T][B][4H]
  float*          o1   = (float*)alloc((size_t)B_ * T_ * D2_ * 4);
  unsigned short* o1bf = (unsigned short*)alloc((size_t)B_ * T_ * D2_ * 2);
  float*          o2   = (float*)alloc((size_t)B_ * T_ * D2_ * 4);
  float*          cst  = (float*)alloc((size_t)2 * B_ * H_ * 4);
  float*          gtv  = (float*)alloc((size_t)B_ * T_ * 4);
  float*          hs2b = (float*)alloc((size_t)B_ * T_ * D2_ * 4);
  float*          part = (float*)alloc((size_t)B_ * T_ * 4);

  auto cvt = [&](const float* src, unsigned short* dst, int rows, int ks, int kd) {
    const int n = rows * kd;
    k_cvt_pad<<<(n + 255) / 256, 256, 0, stream>>>(src, dst, rows, ks, kd);
  };
  cvt(l1Wih[0], w1ih,                      G_, E_,  EP_);
  cvt(l1Wih[1], w1ih + (size_t)G_ * EP_,   G_, E_,  EP_);
  cvt(l1Whh[0], w1hh,                      G_, H_,  H_);
  cvt(l1Whh[1], w1hh + (size_t)G_ * H_,    G_, H_,  H_);
  cvt(l2Wih[0], w2ih,                      G_, D2_, D2_);
  cvt(l2Wih[1], w2ih + (size_t)G_ * D2_,   G_, D2_, D2_);
  cvt(l2Whh[0], w2hh,                      G_, H_,  H_);
  cvt(l2Whh[1], w2hh + (size_t)G_ * H_,    G_, H_,  H_);

  k_embed<<<B_ * T_, 128, 0, stream>>>(ids, wemb, ebuf);

  const int waves = (B_ * T_ / 16) * (G_ / 16);
  const int gblk  = (waves + 7) / 8;
  for (int d = 0; d < 2; ++d)
    k_gemm_bf16<<<gblk, 256, 0, stream>>>(ebuf, w1ih + (size_t)d * G_ * EP_, l1b[d],
                                          xg1 + (size_t)d * B_ * T_ * G_,
                                          B_ * T_, G_, EP_, /*swap_bt=*/1);
  k_lstm_scan<<<16, 256, SCAN_LDS_BYTES, stream>>>(xg1, w1hh, h01, c01, cst, o1);

  cvt(o1, o1bf, B_ * T_, D2_, D2_);
  for (int d = 0; d < 2; ++d)
    k_gemm_bf16<<<gblk, 256, 0, stream>>>(o1bf, w2ih + (size_t)d * G_ * D2_, l2b[d],
                                          xg2 + (size_t)d * B_ * T_ * G_,
                                          B_ * T_, G_, D2_, /*swap_bt=*/1);
  k_lstm_scan<<<16, 256, SCAN_LDS_BYTES, stream>>>(xg2, w2hh, h02, c02, cst, o2);

  k_gt<<<B_ * T_ / 8, 256, 0, stream>>>(o1, gtW, gtb, gtv);
  k_hs2<<<(B_ * D2_ + 255) / 256, 256, 0, stream>>>(o2, gtv, hs2b);
  k_head<<<B_ * T_ / 8, 256, 0, stream>>>(o1, hs2b, c1W, c1b, c2W, c2b, c3W, c3b,
                                          trans, yop, yb, yoe,
                                          (float*)d_out, part);
  k_loss<<<1, 256, 0, stream>>>(part, (float*)d_out + B_ * T_);
}
